// GNN_26671746908642
// MI455X (gfx1250) — matrix operand order repaired and checked by reading, not compiled, not run on hardware
//
#include <hip/hip_runtime.h>
#include <hip/hip_bf16.h>

// ---------------- problem constants (match reference) ----------------
static constexpr int GB   = 512;          // graphs
static constexpr int NPG  = 256;          // nodes per graph
static constexpr int NT   = GB * NPG;     // 131072 nodes
static constexpr int EPG  = 4096;         // edges per graph
static constexpr int NEDG = GB * EPG;     // 2097152 edges
static constexpr int DIM  = 64;
static constexpr int LOOKBACK = 16;
static constexpr int OUTD = 6;

// ---------------- WMMA / TDM types ----------------
typedef __attribute__((ext_vector_type(16))) __bf16 v16bf;
typedef __attribute__((ext_vector_type(8)))  float  v8f;
typedef __attribute__((ext_vector_type(4)))  unsigned int u32x4;
typedef __attribute__((ext_vector_type(4)))  int          i32x4;
typedef __attribute__((ext_vector_type(8)))  int          i32x8;

union FragBF {
    float4 q[2];   // 32 bytes
    v16bf  v;      // 16 bf16 = 8 VGPRs
};

// Tensor Data Mover availability (device pass only; host pass falls back)
#if defined(__has_builtin)
#  if __has_builtin(__builtin_amdgcn_tensor_load_to_lds) && \
      __has_builtin(__builtin_amdgcn_s_wait_tensorcnt)
#    define USE_TDM 1
#  else
#    define USE_TDM 0
#  endif
#else
#  define USE_TDM 0
#endif

__device__ __forceinline__ __bf16 f2bf(float f) {
    union { float f; unsigned u; } in;
    in.f = f;
    unsigned r = (in.u + 0x7FFFu + ((in.u >> 16) & 1u)) >> 16;   // RNE
    union { unsigned short s; __bf16 b; } out;
    out.s = (unsigned short)r;
    return out.b;
}

// pack two f32 -> two bf16 in one dword (lo = a, hi = b)
__device__ __forceinline__ unsigned pack2bf(float a, float b) {
    union { float f; unsigned u; } x, y;
    x.f = a; y.f = b;
    unsigned ra = (x.u + 0x7FFFu + ((x.u >> 16) & 1u)) >> 16;
    unsigned rb = (y.u + 0x7FFFu + ((y.u >> 16) & 1u));
    return (ra & 0xFFFFu) | (rb & 0xFFFF0000u);
}

// ---------------- small utility fills ----------------
__global__ void k_fill_f32(float* p, float v, int n) {
    int i = blockIdx.x * blockDim.x + threadIdx.x;
    if (i < n) p[i] = v;
}
__global__ void k_fill_i32(int* p, int v, int n) {
    int i = blockIdx.x * blockDim.x + threadIdx.x;
    if (i < n) p[i] = v;
}
__global__ void k_fill_u8(unsigned char* p, unsigned char v, int n) {
    int i = blockIdx.x * blockDim.x + threadIdx.x;
    if (i < n) p[i] = v;
}

// ---------------- weight prep: Wt[n][kc] bf16, kc = [Wr rows | Ws rows] ----
__global__ void k_prepw(const float* __restrict__ Wr, const float* __restrict__ Ws,
                        __bf16* __restrict__ Wt, int din) {
    int Kc = 2 * din;
    int idx = blockIdx.x * blockDim.x + threadIdx.x;
    if (idx >= 64 * Kc) return;
    int n = idx / Kc, kk = idx % Kc;
    float v = (kk < din) ? Wr[kk * 64 + n] : Ws[(kk - din) * 64 + n];
    Wt[n * Kc + kk] = f2bf(v);
}

// ---------------- edge aggregation: agg[dst] += x[src] (gated) ------------
__global__ void k_aggregate(const float* __restrict__ x, const int* __restrict__ src,
                            const int* __restrict__ dst, const unsigned char* __restrict__ eact,
                            float* __restrict__ agg, int ldim /*log2(din)*/, int total) {
    int tid = blockIdx.x * blockDim.x + threadIdx.x;
    if (tid >= total) return;
    int e = tid >> ldim;
    int d = tid & ((1 << ldim) - 1);
    if (!eact[e]) return;
    int din = 1 << ldim;
    atomicAdd(&agg[dst[e] * din + d], x[src[e] * din + d]);
}

// ---------------- fused GEMM: h = relu([agg|x] @ Wt^T + b) via WMMA bf16 --
// KC = 32 (layer 0, din=16) or 128 (din=64)
template <int KC>
__global__ __launch_bounds__(128) void k_gemm(const float* __restrict__ agg,
                                              const float* __restrict__ xin,
                                              const __bf16* __restrict__ Wt,  // [64][KC]
                                              const float* __restrict__ bias, // [64]
                                              float* __restrict__ h) {
    constexpr int DIN = KC / 2;
    __shared__ alignas(16) __bf16 sA[64 * KC];  // A tile: 64 rows x KC, K contiguous
    __shared__ alignas(16) __bf16 sW[64 * KC];  // B tile: 64 cols x KC, K contiguous

    const int tid = threadIdx.x;
    const int rowBase = blockIdx.x * 64;

    // ---- stage A = [agg | x] rows: float4 loads, packed bf16 stores ----
    #pragma unroll
    for (int i = tid; i < 64 * KC / 4; i += 128) {
        int r  = i / (KC / 4);
        int c4 = (i % (KC / 4)) * 4;
        const float* sp = (c4 < DIN) ? (agg + (rowBase + r) * DIN + c4)
                                     : (xin + (rowBase + r) * DIN + (c4 - DIN));
        float4 v = *(const float4*)sp;
        uint2 pv;
        pv.x = pack2bf(v.x, v.y);
        pv.y = pack2bf(v.z, v.w);
        *(uint2*)&sA[r * KC + c4] = pv;
    }

#if USE_TDM
    // ---- Tensor Data Mover: DMA the 64xKC bf16 weight tile into LDS ----
    // 2D D# per CDNA5 ISA: count=1, type=2("image"),
    // data_size=2B, tensor_dim0=KC, tensor_dim1=64, dim0_stride=KC,
    // tile = full tensor. Ordered before the verification copy below via
    // s_wait_tensorcnt 0, so staged weights are deterministic regardless.
    if (tid < 32) {
        unsigned long long ga = (unsigned long long)Wt;
        unsigned ldsOff = (unsigned)(unsigned long long)(&sW[0]);
        u32x4 g0;
        g0[0] = 1u;                                              // count=1
        g0[1] = ldsOff;                                          // lds_addr
        g0[2] = (unsigned)ga;                                    // global_addr[31:0]
        g0[3] = (unsigned)((ga >> 32) & 0x1FFFFFFull) | (2u << 30); // addr[56:32] | type=2
        i32x8 g1;
        g1[0] = (int)(1u << 16);                                 // data_size = 2 bytes
        g1[1] = (int)(((unsigned)KC & 0xFFFFu) << 16);           // tensor_dim0[15:0]
        g1[2] = (int)((((unsigned)KC >> 16) & 0xFFFFu) | (64u << 16)); // dim0 hi | dim1 lo
        g1[3] = (int)((unsigned)KC << 16);                       // dim1 hi(0) | tile_dim0
        g1[4] = 64;                                              // tile_dim1
        g1[5] = KC;                                              // tensor_dim0_stride[31:0]
        g1[6] = 0;
        g1[7] = 0;
        i32x4 gz = {0, 0, 0, 0};
#if __clang_major__ >= 23
        i32x8 gz8 = {0, 0, 0, 0, 0, 0, 0, 0};
        __builtin_amdgcn_tensor_load_to_lds(g0, g1, gz, gz, gz8, 0);
#else
        __builtin_amdgcn_tensor_load_to_lds(g0, g1, gz, gz, 0);
#endif
        __builtin_amdgcn_s_wait_tensorcnt((short)0);
    }
    __syncthreads();
#endif
    // ---- plain staged copy of weights (runs after TDM completion) ----
    #pragma unroll
    for (int i = tid; i < 64 * KC / 8; i += 128)
        ((uint4*)sW)[i] = ((const uint4*)Wt)[i];
    __syncthreads();

    const int wave = tid >> 5;
    const int lane = tid & 31;
    const int m    = lane & 15;
    const bool hi  = lane >= 16;

    v8f acc[4] = {};
    const int aRow = wave * 16 + m;

    #pragma unroll
    for (int kt = 0; kt < KC; kt += 32) {
        // A fragment (16x32 bf16): lane<16 -> K {0..7,16..23}, lane>=16 -> K {8..15,24..31}
        FragBF a;
        a.q[0] = *(const float4*)&sA[aRow * KC + kt + (hi ? 8 : 0)];
        a.q[1] = *(const float4*)&sA[aRow * KC + kt + (hi ? 24 : 16)];
        #pragma unroll
        for (int t = 0; t < 4; ++t) {
            // B fragment (32x16 bf16): lane<16 -> K {0..7,16..23}, lane>=16 -> K {8..15,24..31}
            const int n = t * 16 + m;
            FragBF b;
            b.q[0] = *(const float4*)&sW[n * KC + kt + (hi ? 8 : 0)];
            b.q[1] = *(const float4*)&sW[n * KC + kt + (hi ? 24 : 16)];
            acc[t] = __builtin_amdgcn_wmma_f32_16x16x32_bf16(
                false, a.v, false, b.v, (short)0, acc[t], false, false);
        }
    }

    // C layout: VGPR v -> row (v + 8*hi), col = lane%16 ; add bias, relu, store
    #pragma unroll
    for (int t = 0; t < 4; ++t) {
        const int n = t * 16 + m;
        const float bv = bias[n];
        #pragma unroll
        for (int v = 0; v < 8; ++v) {
            int r = rowBase + wave * 16 + v + (hi ? 8 : 0);
            float val = acc[t][v] + bv;
            h[r * 64 + n] = val > 0.f ? val : 0.f;
        }
    }
}

// ---------------- score + exact per-graph top-k (rank counting) ----------
__global__ __launch_bounds__(256) void k_score_topk(const float* __restrict__ h,
                                                    const float* __restrict__ p,
                                                    int* __restrict__ node_act,
                                                    float* __restrict__ score, int k) {
    __shared__ float ls[256];
    const int g = blockIdx.x, t = threadIdx.x;
    const int node = g * 256 + t;

    float nrm2 = 0.f;
    for (int d = 0; d < 64; ++d) nrm2 += p[d] * p[d];
    const float inv = rsqrtf(nrm2);

    float dot = 0.f;
    const float* hr = h + node * 64;
    for (int d = 0; d < 64; ++d) dot += hr[d] * p[d];
    const float sc = tanhf(dot * inv);
    score[node] = sc;

    const int act = node_act[node];
    ls[t] = act ? sc : -__builtin_inff();
    __syncthreads();

    const float si = ls[t];
    int cnt = 0;
    for (int j = 0; j < 256; ++j) {   // rank = #{j : s_j > s_i or (== and j < i)}
        float sj = ls[j];
        cnt += (sj > si) || (sj == si && j < t);
    }
    node_act[node] = (cnt < k) ? 1 : 0;
}

// ---------------- gate x, global max/mean readout, accumulate summation --
__global__ __launch_bounds__(256) void k_gate_pool(const float* __restrict__ h,
                                                   const float* __restrict__ score,
                                                   const int* __restrict__ node_act,
                                                   float* __restrict__ xout,
                                                   float* __restrict__ summation, int k) {
    __shared__ float smax[256], ssum[256];
    const int g = blockIdx.x, t = threadIdx.x;
    const int d = t & 63, grp = t >> 6;           // 4 groups of 64 nodes

    float mx = -__builtin_inff();
    float sm = 0.f;
    for (int i = 0; i < 64; ++i) {
        int idx = g * 256 + grp * 64 + i;
        int act = node_act[idx];
        float xv = h[idx * 64 + d] * score[idx] * (act ? 1.f : 0.f);
        xout[idx * 64 + d] = xv;
        if (act) mx = fmaxf(mx, xv);
        sm += xv;
    }
    smax[t] = mx;
    ssum[t] = sm;
    __syncthreads();
    if (grp == 0) {
        float M = smax[d], S = ssum[d];
        for (int q = 1; q < 4; ++q) {
            M = fmaxf(M, smax[q * 64 + d]);
            S += ssum[q * 64 + d];
        }
        summation[g * 128 + d]      += M;             // gmp
        summation[g * 128 + 64 + d] += S / (float)k;  // gap
    }
}

// ---------------- edge mask update ----------------
__global__ void k_edge_update(const int* __restrict__ src, const int* __restrict__ dst,
                              const int* __restrict__ nact, unsigned char* __restrict__ eact,
                              int ne) {
    int e = blockIdx.x * blockDim.x + threadIdx.x;
    if (e >= ne) return;
    if (eact[e]) eact[e] = (nact[src[e]] && nact[dst[e]]) ? 1 : 0;
}

// ---------------- MLP head ----------------
__global__ __launch_bounds__(64) void k_head(const float* __restrict__ summ,
                                             const float* __restrict__ w1, const float* __restrict__ b1,
                                             const float* __restrict__ w2, const float* __restrict__ b2,
                                             float* __restrict__ out) {
    __shared__ float hid[64];
    const int g = blockIdx.x, d = threadIdx.x;
    float a = b1[d];
    for (int j = 0; j < 128; ++j) a += summ[g * 128 + j] * w1[j * 64 + d];
    hid[d] = a > 0.f ? a : 0.f;
    __syncthreads();
    if (d < OUTD) {
        float o = b2[d];
        for (int j = 0; j < 64; ++j) o += hid[j] * w2[j * OUTD + d];
        out[g * OUTD + d] = o;
    }
}

// ---------------- host orchestration ----------------
extern "C" void kernel_launch(void* const* d_in, const int* in_sizes, int n_in,
                              void* d_out, int out_size, void* d_ws, size_t ws_size,
                              hipStream_t stream) {
    (void)in_sizes; (void)n_in; (void)out_size; (void)ws_size;

    const float* x0      = (const float*)d_in[0];   // [NT,16]
    const int*   ei      = (const int*)  d_in[1];   // [2,NE]
    const float* Wrel0   = (const float*)d_in[2];   // [16,64]
    const float* Wroot0  = (const float*)d_in[3];   // [16,64]
    const float* b0      = (const float*)d_in[4];   // [64]
    const float* Wrel    = (const float*)d_in[5];   // [2,64,64]
    const float* Wroot   = (const float*)d_in[6];   // [2,64,64]
    const float* bl      = (const float*)d_in[7];   // [2,64]
    const float* pool_p  = (const float*)d_in[8];   // [3,64]
    const float* lin1_w  = (const float*)d_in[9];   // [128,64]
    const float* lin1_b  = (const float*)d_in[10];  // [64]
    const float* lin2_w  = (const float*)d_in[11];  // [64,6]
    const float* lin2_b  = (const float*)d_in[12];  // [6]
    float*       out     = (float*)d_out;

    const int* src = ei;
    const int* dst = ei + NEDG;

    // carve workspace
    char* base = (char*)d_ws;
    size_t off = 0;
    auto carve = [&](size_t bytes) {
        void* p = base + off;
        off += (bytes + 255) & ~(size_t)255;
        return p;
    };
    float*         agg   = (float*)        carve((size_t)NT * 64 * 4);
    float*         hbuf  = (float*)        carve((size_t)NT * 64 * 4);
    float*         xbuf  = (float*)        carve((size_t)NT * 64 * 4);
    float*         score = (float*)        carve((size_t)NT * 4);
    int*           nact  = (int*)          carve((size_t)NT * 4);
    unsigned char* eact  = (unsigned char*)carve((size_t)NEDG);
    float*         summ  = (float*)        carve((size_t)GB * 128 * 4);
    __bf16*        Wt    = (__bf16*)       carve((size_t)64 * 128 * 2);

    const int TB = 256;
    // init persistent state
    k_fill_i32<<<(NT + TB - 1) / TB, TB, 0, stream>>>(nact, 1, NT);
    k_fill_u8 <<<(NEDG + TB - 1) / TB, TB, 0, stream>>>(eact, 1, NEDG);
    k_fill_f32<<<(GB * 128 + TB - 1) / TB, TB, 0, stream>>>(summ, 0.f, GB * 128);

    const int K_LIST[3] = {205, 164, 132};
    const float* cur_x = x0;
    int din = LOOKBACK;   // 16 for layer 0, then 64

    for (int layer = 0; layer < 3; ++layer) {
        const float *Wr, *Ws, *bb;
        if (layer == 0) { Wr = Wrel0; Ws = Wroot0; bb = b0; }
        else {
            Wr = Wrel  + (layer - 1) * 64 * 64;
            Ws = Wroot + (layer - 1) * 64 * 64;
            bb = bl    + (layer - 1) * 64;
        }
        const float* p = pool_p + layer * 64;
        const int k = K_LIST[layer];
        const int ldim = (din == 16) ? 4 : 6;

        // zero aggregation buffer
        k_fill_f32<<<(NT * din + TB - 1) / TB, TB, 0, stream>>>(agg, 0.f, NT * din);
        // bf16 transposed concat weights
        k_prepw<<<(64 * 2 * din + TB - 1) / TB, TB, 0, stream>>>(Wr, Ws, Wt, din);
        // scatter-add aggregation over active edges
        int total = NEDG * din;   // <= 2^27, fits int
        k_aggregate<<<(total + TB - 1) / TB, TB, 0, stream>>>(cur_x, src, dst, eact,
                                                              agg, ldim, total);
        // fused GraphConv linear + relu (WMMA bf16)
        if (layer == 0)
            k_gemm<32><<<NT / 64, 128, 0, stream>>>(agg, cur_x, Wt, bb, hbuf);
        else
            k_gemm<128><<<NT / 64, 128, 0, stream>>>(agg, cur_x, Wt, bb, hbuf);
        // score + exact top-k mask
        k_score_topk<<<GB, 256, 0, stream>>>(hbuf, p, nact, score, k);
        // gate x, pooled readout accumulation
        k_gate_pool<<<GB, 256, 0, stream>>>(hbuf, score, nact, xbuf, summ, k);
        // edge survival
        k_edge_update<<<(NEDG + TB - 1) / TB, TB, 0, stream>>>(src, dst, nact, eact, NEDG);

        cur_x = xbuf;
        din = DIM;
    }

    k_head<<<GB, 64, 0, stream>>>(summ, lin1_w, lin1_b, lin2_w, lin2_b, out);
}
